// second_order_RNN_84232898609620
// MI455X (gfx1250) — matrix operand order repaired and checked by reading, not compiled, not run on hardware
//
#include <hip/hip_runtime.h>
#include <hip/hip_bf16.h>

typedef __bf16 v16bf __attribute__((ext_vector_type(16)));
typedef float  v8f   __attribute__((ext_vector_type(8)));

#define Bsz 512
#define Dd  128
#define Ll  128
#define Rr  256
#define Oo  128
#define KSPLIT 4
#define RCHUNK 64                 // Rr / KSPLIT
#define HPART_ELEMS (Bsz*Rr)      // 131072 floats per K-split partial

static __device__ __forceinline__ v8f wmma_bf16(v16bf a, v16bf b, v8f c) {
    return __builtin_amdgcn_wmma_f32_16x16x32_bf16(false, a, false, b,
                                                   (short)0, c, false, false);
}

// ---------------- prep: pack transition T[r][d][s] (fp32) into bf16 WMMA
// B-fragments. Fragment (r, db, nt) is the 32(K=d)x16(N=s) tile for
// d in [db*32, db*32+32), s in [nt*16, nt*16+16).
// B-frag lane layout (16-bit): col n = lane&15;
//   e<8 : K = (lane<16?0:8)+e ; e>=8 : K = 16+(lane<16?0:8)+(e-8).
__global__ __launch_bounds__(256)
void pack_T_kernel(const float* __restrict__ T, __bf16* __restrict__ Tb) {
    int gid  = blockIdx.x * 256 + threadIdx.x;     // 524288 lane-tasks
    int lane = gid & 31;
    int frag = gid >> 5;                           // ((r*4+db)*16+nt)
    int nt = frag & 15;
    int db = (frag >> 4) & 3;
    int r  = frag >> 6;
    int n    = lane & 15;
    int koff = (lane < 16) ? 0 : 8;
    v16bf vals;
    #pragma unroll
    for (int e = 0; e < 16; ++e) {
        int d = db * 32 + ((e < 8) ? (koff + e) : (16 + koff + (e - 8)));
        vals[e] = (__bf16)T[((size_t)r * Dd + d) * Rr + nt * 16 + n];
    }
    *((v16bf*)Tb + (size_t)frag * 32 + lane) = vals;   // 32B per lane
}

// ---------------- prep: pack x[b][d][t] into bf16 WMMA A-fragments.
// Fragment (t, mt, db) is the 16(M=b)x32(K=d) tile.
// A-frag lane layout (16-bit): row m = lane&15;
//   e<8 : K = (lane<16?0:8)+e ; e>=8 : K = 16+(lane<16?0:8)+(e-8).
__global__ __launch_bounds__(256)
void pack_x_kernel(const float* __restrict__ x, __bf16* __restrict__ xp) {
    int gid  = blockIdx.x * 256 + threadIdx.x;     // 524288 lane-tasks
    int lane = gid & 31;
    int frag = gid >> 5;                           // ((t*32+mt)*4+db)
    int db = frag & 3;
    int mt = (frag >> 2) & 31;
    int t  = frag >> 7;
    int m    = lane & 15;
    int b    = mt * 16 + m;
    int koff = (lane < 16) ? 0 : 8;
    v16bf vals;
    #pragma unroll
    for (int e = 0; e < 16; ++e) {
        int d = db * 32 + ((e < 8) ? (koff + e) : (16 + koff + (e - 8)));
        vals[e] = (__bf16)x[((size_t)b * Dd + d) * Ll + t];
    }
    *((v16bf*)xp + (size_t)frag * 32 + lane) = vals;
}

// ---------------- prep: h0 partials. Chunk 0 = broadcast(alpha), chunks 1..3 = 0.
__global__ __launch_bounds__(256)
void init_h_kernel(const float* __restrict__ alpha, float* __restrict__ hA) {
    int gid = blockIdx.x * 256 + threadIdx.x;      // KSPLIT*B*R = 524288
    int r = gid & (Rr - 1);
    int c = gid >> 17;                             // / (Bsz*Rr)
    hA[gid] = (c == 0) ? alpha[r] : 0.0f;
}

// ---------------- one recurrence step:
//   hdst[rc][b][s] = sum_{r in chunk rc} h[b][r] * (sum_d x_t[b][d] T[r][d][s])
// h[b][r] is reconstructed as the sum of the previous step's 4 partials.
// Grid: 256 blocks = {nt:16} x {rc:4} x {mgroup:4}; 8 waves = 8 mtiles that
// share the same T fragments (WGP$-friendly) and a 32KB LDS h tile.
// Inner loop is software-pipelined: B-fragments and the h scalar for r+1 are
// fetched while the 4-WMMA chain for r executes.
__global__ __launch_bounds__(256)
void rnn_step_kernel(const __bf16* __restrict__ xp, const __bf16* __restrict__ Tb,
                     const float* __restrict__ hsrc, float* __restrict__ hdst,
                     int t) {
    __shared__ float hlds[8 * RCHUNK * 16];        // [mtile_local][r_local][m]

    const int tid  = threadIdx.x;
    const int lane = tid & 31;
    const int wave = tid >> 5;
    const int bid  = blockIdx.x;
    const int nt = bid & 15;
    const int rc = (bid >> 4) & 3;
    const int mg = bid >> 6;
    const int mt = mg * 8 + wave;

    // Cooperative fill of the fp32 h tile (sum of previous K-split partials).
    for (int i = tid; i < 8 * RCHUNK * 16; i += 256) {
        int wl = i >> 10;            // local mtile (RCHUNK*16 = 1024)
        int rl = (i >> 4) & (RCHUNK - 1);
        int m  = i & 15;
        int b  = (mg * 8 + wl) * 16 + m;
        int r  = rc * RCHUNK + rl;
        size_t off = (size_t)b * Rr + r;
        hlds[i] = hsrc[off] + hsrc[HPART_ELEMS + off]
                + hsrc[2 * HPART_ELEMS + off] + hsrc[3 * HPART_ELEMS + off];
    }
    __syncthreads();

    // x A-fragments for this (t, mt): 4 d-blocks, held in VGPRs all step.
    const char* xB = (const char*)xp + (((size_t)(t * 32 + mt) * 4) * 32 + lane) * 32;
    v16bf xf0 = *(const v16bf*)(xB);
    v16bf xf1 = *(const v16bf*)(xB + 1024);
    v16bf xf2 = *(const v16bf*)(xB + 2048);
    v16bf xf3 = *(const v16bf*)(xB + 3072);

    // T B-fragment stream: +16384B per d-block, +65536B per r.
    const char* tB = (const char*)Tb
                   + ((((size_t)(rc * RCHUNK) * 4) * 16 + nt) * 32 + lane) * 32;
    const float* hl = &hlds[wave * 1024 + (lane & 15)];

    // Pipeline prologue: fragments + h scalar for r = 0.
    v16bf b0 = *(const v16bf*)(tB);
    v16bf b1 = *(const v16bf*)(tB + 16384);
    v16bf b2 = *(const v16bf*)(tB + 32768);
    v16bf b3 = *(const v16bf*)(tB + 49152);
    float hs = hl[0];

    v8f C = {};
    for (int rl = 0; rl < RCHUNK - 1; ++rl) {
        // Issue next iteration's loads before this iteration's WMMA chain.
        const char* tN = tB + 65536;
        v16bf n0 = *(const v16bf*)(tN);
        v16bf n1 = *(const v16bf*)(tN + 16384);
        v16bf n2 = *(const v16bf*)(tN + 32768);
        v16bf n3 = *(const v16bf*)(tN + 49152);
        float hn = hl[(rl + 1) * 16];
        __builtin_prefetch(tN + 65536, 0, 3);      // r+2, near-cache hint

        v8f P = {};
        P = wmma_bf16(xf0, b0, P);
        P = wmma_bf16(xf1, b1, P);
        P = wmma_bf16(xf2, b2, P);
        P = wmma_bf16(xf3, b3, P);
        #pragma unroll
        for (int j = 0; j < 8; ++j) C[j] += hs * P[j];   // fp32 h applied post-WMMA

        b0 = n0; b1 = n1; b2 = n2; b3 = n3; hs = hn;
        tB = tN;
    }
    { // epilogue: last r, no further loads
        v8f P = {};
        P = wmma_bf16(xf0, b0, P);
        P = wmma_bf16(xf1, b1, P);
        P = wmma_bf16(xf2, b2, P);
        P = wmma_bf16(xf3, b3, P);
        #pragma unroll
        for (int j = 0; j < 8; ++j) C[j] += hs * P[j];
    }

    // Store the 16x16 fp32 partial tile (C/D layout: vgpr j, lane -> M=j+(lane>>4)*8, N=lane&15).
    const int mhi = (lane >> 4) * 8;
    const int n   = lane & 15;
    float* dp = hdst + (size_t)rc * HPART_ELEMS + ((size_t)mt * 16) * Rr + nt * 16 + n;
    #pragma unroll
    for (int j = 0; j < 8; ++j)
        dp[(size_t)(j + mhi) * Rr] = C[j];
}

// ---------------- final projection: out[b][o] = (sum_c h_part[c][b][:]) @ omega
__global__ __launch_bounds__(128)
void rnn_out_kernel(const float* __restrict__ hsrc, const float* __restrict__ omega,
                    float* __restrict__ out) {
    __shared__ float hrow[Rr];
    int b = blockIdx.x;
    for (int s = threadIdx.x; s < Rr; s += 128) {
        size_t off = (size_t)b * Rr + s;
        hrow[s] = hsrc[off] + hsrc[HPART_ELEMS + off]
                + hsrc[2 * HPART_ELEMS + off] + hsrc[3 * HPART_ELEMS + off];
    }
    __syncthreads();
    int o = threadIdx.x;
    float acc = 0.0f;
    for (int s = 0; s < Rr; ++s)
        acc = fmaf(hrow[s], omega[(size_t)s * Oo + o], acc);
    out[(size_t)b * Oo + o] = acc;
}

extern "C" void kernel_launch(void* const* d_in, const int* in_sizes, int n_in,
                              void* d_out, int out_size, void* d_ws, size_t ws_size,
                              hipStream_t stream) {
    const float* x     = (const float*)d_in[0];   // [512,128,128]
    const float* alpha = (const float*)d_in[1];   // [256]
    const float* T     = (const float*)d_in[2];   // [256,128,256]
    const float* omega = (const float*)d_in[3];   // [256,128]
    float* out = (float*)d_out;                   // [512,128]

    char* ws = (char*)d_ws;
    __bf16* Tb = (__bf16*)(ws + 0);               // 16,777,216 B
    __bf16* xp = (__bf16*)(ws + 16777216);        // 16,777,216 B
    float*  hA = (float*)(ws + 33554432);         //  2,097,152 B (4 partials)
    float*  hB = (float*)(ws + 35651584);         //  2,097,152 B
    // total ws used: 37,748,736 B

    pack_T_kernel<<<2048, 256, 0, stream>>>(T, Tb);
    pack_x_kernel<<<2048, 256, 0, stream>>>(x, xp);
    init_h_kernel<<<2048, 256, 0, stream>>>(alpha, hA);

    for (int t = 0; t < Ll; ++t) {
        const float* src = (t & 1) ? hB : hA;
        float*       dst = (t & 1) ? hA : hB;
        rnn_step_kernel<<<256, 256, 0, stream>>>(xp, Tb, src, dst, t);
    }
    // L=128 is even -> final h partials live in hA
    rnn_out_kernel<<<Bsz, 128, 0, stream>>>(hA, omega, out);
}
